// UNO_54022098649904
// MI455X (gfx1250) — compile-verified, matrix operand order checked by reading, not run on hardware
//
#include <hip/hip_runtime.h>
#include <math.h>

#define MODES 12
#define NXDIM 512
#define CIN0  10

typedef __attribute__((ext_vector_type(16))) _Float16 v16h;
typedef __attribute__((ext_vector_type(8)))  float    v8f;

__device__ __forceinline__ float gelu_t(float x) {
  float x3 = x * x * x;
  return 0.5f * x * (1.0f + tanhf(0.7978845608028654f * (x + 0.044715f * x3)));
}

union HFrag { uint4 q[2]; v16h v; };

// ---------------------------------------------------------------------------
// C[M,N] (f32) = A[M,Kp] (f16, K zero-padded to mult of 32) x BT[N,Kp]^T (f16)
// + bias[N];  actmode==1 -> fused tanh-GELU.
// One wave per 16x16 C tile, 4 waves per block along M.
// A-fragment per ISA 16-bit 16x32 layout: lane L (m = L&15) holds two 16B
// chunks at K offsets { (L>=16?8:0), +16 }. BT rows load identically (B is
// "A-like" when stored as N x K).
// ---------------------------------------------------------------------------
__global__ void gemm_f16_wmma(const _Float16* __restrict__ A,
                              const _Float16* __restrict__ BT,
                              const float* __restrict__ bias,
                              float* __restrict__ C,
                              int M, int N, int Kp, int actmode)
{
  const int lane = threadIdx.x & 31;
  const int wave = threadIdx.x >> 5;
  const int m0 = (blockIdx.x * 4 + wave) * 16;
  const int n0 = blockIdx.y * 16;
  if (m0 >= M) return;                       // wave-uniform: EXEC stays all-1s
  int mr = m0 + (lane & 15); if (mr >= M) mr = M - 1;
  int nr = n0 + (lane & 15); if (nr >= N) nr = N - 1;
  const int kh = (lane >> 4) << 3;           // 0 or 8
  const _Float16* Ap = A  + (size_t)mr * Kp + kh;
  const _Float16* Bp = BT + (size_t)nr * Kp + kh;
  v8f acc = {0.f, 0.f, 0.f, 0.f, 0.f, 0.f, 0.f, 0.f};
  for (int k0 = 0; k0 < Kp; k0 += 32) {
    HFrag fa, fb;
    fa.q[0] = *(const uint4*)(Ap + k0);
    fa.q[1] = *(const uint4*)(Ap + k0 + 16);
    fb.q[0] = *(const uint4*)(Bp + k0);
    fb.q[1] = *(const uint4*)(Bp + k0 + 16);
    acc = __builtin_amdgcn_wmma_f32_16x16x32_f16(false, fa.v, false, fb.v,
                                                 (short)0, acc, false, false);
  }
  const int n = n0 + (lane & 15);
  if (n < N) {
    const float bb = bias ? bias[n] : 0.f;
    const int mbase = m0 + ((lane >> 4) << 3);
#pragma unroll
    for (int j = 0; j < 8; ++j) {
      int m = mbase + j;
      if (m < M) {
        float v = acc[j] + bb;
        if (actmode == 1) v = gelu_t(v);
        C[(size_t)m * N + n] = v;
      }
    }
  }
}

// ---- elementwise / staging kernels ----------------------------------------

// Build fc0 input: [x(10ch), gridx, gridy, 0...] as f16, Kp=32.
__global__ void build_fc0_in(const float* __restrict__ x, _Float16* __restrict__ dst, int total)
{
  int i = blockIdx.x * blockDim.x + threadIdx.x;
  if (i >= total) return;
  int p = i >> 5;
  int c = i & 31;
  int xi = p / NXDIM, yj = p - xi * NXDIM;
  float v;
  if (c < CIN0)           v = x[(size_t)p * CIN0 + c];
  else if (c == CIN0)     v = (float)yj * (1.0f / (NXDIM - 1));
  else if (c == CIN0 + 1) v = (float)xi * (1.0f / (NXDIM - 1));
  else                    v = 0.f;
  dst[i] = (_Float16)v;
}

// f32 (M,C) -> f16 (M,Kp) with zero pad.
__global__ void f32_to_f16_pad(const float* __restrict__ src, _Float16* __restrict__ dst,
                               int C, int Kp, int total)
{
  int i = blockIdx.x * blockDim.x + threadIdx.x;
  if (i >= total) return;
  int m = i / Kp, c = i - m * Kp;
  dst[i] = (c < C) ? (_Float16)src[(size_t)m * C + c] : (_Float16)0.f;
}

// W (K,N) f32 -> WT (N,Kp) f16 (transposed, zero K-pad).
__global__ void wconv_f16(const float* __restrict__ W, _Float16* __restrict__ WT,
                          int K, int N, int Kp, int total)
{
  int i = blockIdx.x * blockDim.x + threadIdx.x;
  if (i >= total) return;
  int n = i / Kp, k = i - n * Kp;
  WT[i] = (k < K) ? (_Float16)W[(size_t)k * N + n] : (_Float16)0.f;
}

// channel concat: dst (M, Ca+Cb) = [a | b]
__global__ void concat2(const float* __restrict__ a, const float* __restrict__ b,
                        float* __restrict__ dst, int Ca, int Cb, int total)
{
  int i = blockIdx.x * blockDim.x + threadIdx.x;
  if (i >= total) return;
  int Ct = Ca + Cb;
  int p = i / Ct, c = i - p * Ct;
  dst[i] = (c < Ca) ? a[(size_t)p * Ca + c] : b[(size_t)p * Cb + (c - Ca)];
}

// ---- partial-DFT spectral path --------------------------------------------

// Stage 1 forward: Y[x][ky][c] = sum_y act[x,y,c] * e^{-2pi i ky y / s}, ky<12
__global__ void dft_y(const float* __restrict__ act, float* __restrict__ Y,
                      int s, int C, int total)
{
  int i = blockIdx.x * blockDim.x + threadIdx.x;
  if (i >= total) return;
  int c = i % C; int t = i / C; int ky = t % MODES; int xr = t / MODES;
  float step = -6.283185307f * (float)ky / (float)s;
  float cs = cosf(step), sn = sinf(step);
  float cr = 1.f, ci = 0.f, ar = 0.f, ai = 0.f;
  const float* row = act + (size_t)xr * s * C + c;
  for (int y = 0; y < s; ++y) {
    float a = row[(size_t)y * C];
    ar += a * cr; ai += a * ci;
    float tt = cr * cs - ci * sn; ci = cr * sn + ci * cs; cr = tt;
  }
  size_t o = ((size_t)(xr * MODES + ky) * C + c) * 2;
  Y[o] = ar; Y[o + 1] = ai;
}

// Stage 2 forward: F[t][ky][c], t<12 -> kx=t; t>=12 -> row s-12+(t-12).
__global__ void dft_x(const float* __restrict__ Y, float* __restrict__ F,
                      int s, int C, int total)
{
  int i = blockIdx.x * blockDim.x + threadIdx.x;
  if (i >= total) return;
  int c = i % C; int t2 = i / C; int ky = t2 % MODES; int t = t2 / MODES;
  int r = (t < MODES) ? t : (s - 2 * MODES + t);
  float step = -6.283185307f * (float)r / (float)s;
  float cs = cosf(step), sn = sinf(step);
  float cr = 1.f, ci = 0.f, ar = 0.f, ai = 0.f;
  for (int x = 0; x < s; ++x) {
    size_t yo = ((size_t)(x * MODES + ky) * C + c) * 2;
    float yr = Y[yo], yi = Y[yo + 1];
    ar += yr * cr - yi * ci;
    ai += yr * ci + yi * cr;
    float tt = cr * cs - ci * sn; ci = cr * sn + ci * cs; cr = tt;
  }
  size_t o = ((size_t)(t * MODES + ky) * C + c) * 2;
  F[o] = ar; F[o + 1] = ai;
}

// Per-mode channel mix; real/imag multiplied SEPARATELY (faithful to ref).
__global__ void mode_mix(const float* __restrict__ F, const float* __restrict__ w1,
                         const float* __restrict__ w2, float* __restrict__ MX,
                         int Cin, int Cout, int total)
{
  int i = blockIdx.x * blockDim.x + threadIdx.x;
  if (i >= total) return;
  int o = i % Cout; int t2 = i / Cout; int ky = t2 % MODES; int t = t2 / MODES;
  const float* w = (t < MODES) ? w1 : w2;
  int kx = (t < MODES) ? t : t - MODES;
  float ar = 0.f, ai = 0.f;
  const float* fp = F + (size_t)(t * MODES + ky) * Cin * 2;
  for (int ic = 0; ic < Cin; ++ic) {
    size_t wi = ((((size_t)ic * MODES + kx) * MODES + ky) * Cout + o) * 2;
    ar += fp[ic * 2]     * w[wi];
    ai += fp[ic * 2 + 1] * w[wi + 1];
  }
  size_t oo = ((size_t)(t * MODES + ky) * Cout + o) * 2;
  MX[oo] = ar; MX[oo + 1] = ai;
}

// Inverse stage over x: modes with row index r >= so are cropped (numpy ifft(n<len)).
__global__ void idft_x(const float* __restrict__ MX, float* __restrict__ G,
                       int s, int so, int Cout, int total)
{
  int i = blockIdx.x * blockDim.x + threadIdx.x;
  if (i >= total) return;
  int o = i % Cout; int t2 = i / Cout; int ky = t2 % MODES; int xp = t2 / MODES;
  float ar = 0.f, ai = 0.f;
  for (int t = 0; t < 2 * MODES; ++t) {
    int r = (t < MODES) ? t : (s - 2 * MODES + t);
    if (r >= so) continue;
    int ph = (int)(((long long)r * xp) % so);
    float ang = 6.283185307f * (float)ph / (float)so;
    float sv, cv; __sincosf(ang, &sv, &cv);
    size_t mo = ((size_t)(t * MODES + ky) * Cout + o) * 2;
    float mr = MX[mo], mi = MX[mo + 1];
    ar += mr * cv - mi * sv;
    ai += mr * sv + mi * cv;
  }
  size_t go = ((size_t)(xp * MODES + ky) * Cout + o) * 2;
  G[go] = ar; G[go + 1] = ai;
}

// Inverse stage over y (real output, Hermitian weights), 1/(so*so) norm.
__global__ void idft_y(const float* __restrict__ G, float* __restrict__ X1,
                       int so, int Cout, int total)
{
  int i = blockIdx.x * blockDim.x + threadIdx.x;
  if (i >= total) return;
  int o = i % Cout; int p = i / Cout;
  int xp = p / so, yp = p - xp * so;
  float step = 6.283185307f * (float)yp / (float)so;
  float cs = cosf(step), sn = sinf(step);
  float cr = 1.f, ci = 0.f, acc = 0.f;
  const float* g = G + (size_t)xp * MODES * Cout * 2 + (size_t)o * 2;
  for (int ky = 0; ky < MODES; ++ky) {
    float gr = g[(size_t)ky * Cout * 2], gi = g[(size_t)ky * Cout * 2 + 1];
    float wgt = (ky == 0) ? 1.f : 2.f;
    acc += wgt * (gr * cr - gi * ci);
    float tt = cr * cs - ci * sn; ci = cr * sn + ci * cs; cr = tt;
  }
  X1[(size_t)p * Cout + o] = acc / ((float)so * (float)so);
}

// Fused: bilinear resample of conv path (s->so, zero beyond far edge, coords
// linspace(0,s,so)), add spectral X1, LayerNorm over channels, tanh-GELU.
__global__ void ln_res_gelu(const float* __restrict__ X1, const float* __restrict__ CONV,
                            const float* __restrict__ lns, const float* __restrict__ lnb,
                            float* __restrict__ out, int s, int so, int C, int total)
{
  int p = blockIdx.x * blockDim.x + threadIdx.x;
  if (p >= total) return;
  int xp = p / so, yp = p - xp * so;
  float sc = (float)s / (float)(so - 1);
  float px = xp * sc, py = yp * sc;
  int ix = (int)px, iy = (int)py;
  float fx = px - ix, fy = py - iy;
  float w00 = (1 - fx) * (1 - fy), w01 = (1 - fx) * fy;
  float w10 = fx * (1 - fy),       w11 = fx * fy;
  bool vx0 = ix < s, vx1 = (ix + 1) < s, vy0 = iy < s, vy1 = (iy + 1) < s;
  const float* c00 = (vx0 && vy0) ? CONV + ((size_t)ix * s + iy) * C : nullptr;
  const float* c01 = (vx0 && vy1) ? CONV + ((size_t)ix * s + iy + 1) * C : nullptr;
  const float* c10 = (vx1 && vy0) ? CONV + ((size_t)(ix + 1) * s + iy) * C : nullptr;
  const float* c11 = (vx1 && vy1) ? CONV + ((size_t)(ix + 1) * s + iy + 1) * C : nullptr;
  const float* x1p = X1 + (size_t)p * C;
  float mean = 0.f, m2 = 0.f;
  for (int c = 0; c < C; ++c) {
    float v = x1p[c];
    if (c00) v += w00 * c00[c];
    if (c01) v += w01 * c01[c];
    if (c10) v += w10 * c10[c];
    if (c11) v += w11 * c11[c];
    mean += v; m2 += v * v;
  }
  mean /= C;
  float var = m2 / C - mean * mean;
  float inv = rsqrtf(var + 1e-6f);
  float* op = out + (size_t)p * C;
  for (int c = 0; c < C; ++c) {
    float v = x1p[c];
    if (c00) v += w00 * c00[c];
    if (c01) v += w01 * c01[c];
    if (c10) v += w10 * c10[c];
    if (c11) v += w11 * c11[c];
    float g = (v - mean) * inv * lns[c] + lnb[c];
    op[c] = gelu_t(g);
  }
}

// ---------------------------------------------------------------------------
// Host orchestration
// ---------------------------------------------------------------------------
struct BlockW { const float *w1, *w2, *cw, *cb, *lns, *lnb; };
struct Ws {
  float *ACT, *ACT2, *CONV, *X1, *H0, *H1, *H2, *Y, *F, *MIX, *G;
  _Float16 *A16, *W16;
};

#define L1(kern, n, ...) \
  kern<<<dim3((unsigned)(((n) + 255) / 256)), dim3(256), 0, st>>>(__VA_ARGS__)

static void run_block(const float* in, int s, int C, int so, int Cout,
                      const BlockW& W, float* out, const Ws& ws, hipStream_t st)
{
  int Kp = (C + 31) & ~31;
  int n;
  n = s * MODES * C;          L1(dft_y, n, in, ws.Y, s, C, n);
  n = 2 * MODES * MODES * C;  L1(dft_x, n, ws.Y, ws.F, s, C, n);
  n = 2 * MODES * MODES * Cout; L1(mode_mix, n, ws.F, W.w1, W.w2, ws.MIX, C, Cout, n);
  n = so * MODES * Cout;      L1(idft_x, n, ws.MIX, ws.G, s, so, Cout, n);
  n = so * so * Cout;         L1(idft_y, n, ws.G, ws.X1, so, Cout, n);
  n = s * s * Kp;             L1(f32_to_f16_pad, n, in, ws.A16, C, Kp, n);
  n = Cout * Kp;              L1(wconv_f16, n, W.cw, ws.W16, C, Cout, Kp, n);
  gemm_f16_wmma<<<dim3((s * s + 63) / 64, (Cout + 15) / 16), dim3(128), 0, st>>>(
      ws.A16, ws.W16, W.cb, ws.CONV, s * s, Cout, Kp, 0);
  n = so * so;                L1(ln_res_gelu, n, ws.X1, ws.CONV, W.lns, W.lnb, out, s, so, Cout, n);
}

extern "C" void kernel_launch(void* const* d_in, const int* in_sizes, int n_in,
                              void* d_out, int out_size, void* d_ws, size_t ws_size,
                              hipStream_t stream)
{
  (void)out_size; (void)ws_size;
  hipStream_t st = stream;

  // ---- input mapping (detect flatten order) ----
  const float *X, *fc0w, *fc0b, *fc1w, *fc1b, *fc2w, *fc2b;
  BlockW enc[3], dec[3], proc;
  if (in_sizes[0] == NXDIM * NXDIM * CIN0) {
    // insertion order: x, fc0{w,b}, enc0..2{w1,w2,cw,cb,ln_s,ln_b}, proc, dec0..2, fc1, fc2
    X = (const float*)d_in[0];
    fc0w = (const float*)d_in[1]; fc0b = (const float*)d_in[2];
    auto blk = [&](int b) { BlockW w;
      w.w1 = (const float*)d_in[b];     w.w2 = (const float*)d_in[b + 1];
      w.cw = (const float*)d_in[b + 2]; w.cb = (const float*)d_in[b + 3];
      w.lns = (const float*)d_in[b + 4]; w.lnb = (const float*)d_in[b + 5]; return w; };
    enc[0] = blk(3); enc[1] = blk(9); enc[2] = blk(15);
    proc = blk(21);
    dec[0] = blk(27); dec[1] = blk(33); dec[2] = blk(39);
    fc1w = (const float*)d_in[45]; fc1b = (const float*)d_in[46];
    fc2w = (const float*)d_in[47]; fc2b = (const float*)d_in[48];
  } else {
    // jax sorted-key flatten: dec0..2{cb,cw,ln_b,ln_s,w1,w2}, enc0..2, fc0{b,w},
    // fc1{b,w}, fc2{b,w}, proc, then x
    auto blk = [&](int b) { BlockW w;
      w.cb = (const float*)d_in[b];      w.cw = (const float*)d_in[b + 1];
      w.lnb = (const float*)d_in[b + 2]; w.lns = (const float*)d_in[b + 3];
      w.w1 = (const float*)d_in[b + 4];  w.w2 = (const float*)d_in[b + 5]; return w; };
    dec[0] = blk(0); dec[1] = blk(6); dec[2] = blk(12);
    enc[0] = blk(18); enc[1] = blk(24); enc[2] = blk(30);
    fc0b = (const float*)d_in[36]; fc0w = (const float*)d_in[37];
    fc1b = (const float*)d_in[38]; fc1w = (const float*)d_in[39];
    fc2b = (const float*)d_in[40]; fc2w = (const float*)d_in[41];
    proc = blk(42);
    X = (const float*)d_in[n_in - 1];
  }

  // ---- workspace carve (all 256B aligned) ----
  char* base = (char*)d_ws;
  size_t off = 0;
  auto carve = [&](size_t bytes) {
    void* p = base + off;
    off = (off + bytes + 255) & ~(size_t)255;
    return p;
  };
  Ws ws;
  ws.ACT  = (float*)carve((size_t)14155776 * 4);  // concat buffer (max 384^2 x 96 pad)
  ws.ACT2 = (float*)carve((size_t)8388608 * 4);   // block outputs (max 512^2 x 32)
  ws.CONV = (float*)carve((size_t)12582912 * 4);  // conv GEMM out (max 512^2 x 48)
  ws.X1   = (float*)carve((size_t)8388608 * 4);   // spectral out
  ws.H0   = (float*)carve((size_t)7077888 * 4);   // 384^2 x 48
  ws.H1   = (float*)carve((size_t)5971968 * 4);   // 288^2 x 72
  ws.H2   = (float*)carve((size_t)5038848 * 4);   // 216^2 x 108
  ws.Y    = (float*)carve((size_t)1200000 * 4);
  ws.F    = (float*)carve((size_t)131072 * 4);
  ws.MIX  = (float*)carve((size_t)65536 * 4);
  ws.G    = (float*)carve((size_t)600000 * 4);
  ws.A16  = (_Float16*)carve((size_t)14155776 * 2);
  ws.W16  = (_Float16*)carve((size_t)65536 * 2);

  const int M0 = NXDIM * NXDIM;
  int n;

  // ---- fc0: [x, gridx, gridy] @ (12->32) ----
  n = M0 * 32; L1(build_fc0_in, n, X, ws.A16, n);
  n = 32 * 32; L1(wconv_f16, n, fc0w, ws.W16, 12, 32, 32, n);
  gemm_f16_wmma<<<dim3((M0 + 63) / 64, 2), dim3(128), 0, st>>>(
      ws.A16, ws.W16, fc0b, ws.ACT, M0, 32, 32, 0);

  // ---- encoder: 512->384->288->216 ----
  const int S[4]  = {512, 384, 288, 216};
  const int CH[4] = {32, 48, 72, 108};
  float* hbuf[3] = {ws.H0, ws.H1, ws.H2};
  const float* cur = ws.ACT;
  for (int i = 0; i < 3; ++i) {
    run_block(cur, S[i], CH[i], S[i + 1], CH[i + 1], enc[i], hbuf[i], ws, st);
    cur = hbuf[i];
  }

  // ---- processor (scale 1.0) ----
  run_block(ws.H2, 216, 108, 216, 108, proc, ws.ACT2, ws, st);

  // ---- decoder with skip concats: 216->288->384->512 ----
  const int DS[4] = {216, 288, 384, 512};
  const int DCI[3] = {216, 144, 96};
  const int DCO[3] = {72, 48, 32};
  const float* skip[3] = {ws.H2, ws.H1, ws.H0};
  const int skc[3] = {108, 72, 48};
  const int prvc[3] = {108, 72, 48};
  const float* dprev = ws.ACT2;
  for (int i = 0; i < 3; ++i) {
    int s = DS[i], so = DS[i + 1];
    n = s * s * DCI[i];
    L1(concat2, n, dprev, skip[i], ws.ACT, prvc[i], skc[i], n);
    run_block(ws.ACT, s, DCI[i], so, DCO[i], dec[i], ws.ACT2, ws, st);
    dprev = ws.ACT2;
  }

  // ---- fc1 (32->32, fused GELU) ----
  n = M0 * 32; L1(f32_to_f16_pad, n, ws.ACT2, ws.A16, 32, 32, n);
  n = 32 * 32; L1(wconv_f16, n, fc1w, ws.W16, 32, 32, 32, n);
  gemm_f16_wmma<<<dim3((M0 + 63) / 64, 2), dim3(128), 0, st>>>(
      ws.A16, ws.W16, fc1b, ws.CONV, M0, 32, 32, 1);

  // ---- fc2 (32->1) -> d_out ----
  n = M0 * 32; L1(f32_to_f16_pad, n, ws.CONV, ws.A16, 32, 32, n);
  n = 1 * 32;  L1(wconv_f16, n, fc2w, ws.W16, 32, 1, 32, n);
  gemm_f16_wmma<<<dim3((M0 + 63) / 64, 1), dim3(128), 0, st>>>(
      ws.A16, ws.W16, fc2b, (float*)d_out, M0, 1, 32, 0);
}
